// LSTMNerModel_18013092840194
// MI455X (gfx1250) — matrix operand order repaired and checked by reading, not compiled, not run on hardware
//
#include <hip/hip_runtime.h>
#include <stdint.h>

typedef __attribute__((ext_vector_type(16))) __bf16   v16bf;
typedef __attribute__((ext_vector_type(2)))  __bf16   v2bf;
typedef __attribute__((ext_vector_type(8)))  float    v8f;
typedef __attribute__((ext_vector_type(8)))  uint32_t v8u;
typedef __attribute__((ext_vector_type(4)))  uint32_t v4u;

#define T_LEN 8192
#define EMB   128
#define HID   256
#define GATES 1024   // 4*HID, torch gate order i,f,g,o
#define TAGS  17
#define PITCH 132    // LDS row pitch (dwords): 132%64==4 -> conflict-free, 16B-aligned chunks

// ---------- helpers ----------
__device__ __forceinline__ uint32_t cvt_pk(float a, float b) {
  v2bf v; v[0] = (__bf16)a; v[1] = (__bf16)b;       // hw f32->bf16 (RNE)
  return __builtin_bit_cast(uint32_t, v);
}
__device__ __forceinline__ uint16_t f2bf(float a) {
  __bf16 h = (__bf16)a;
  return __builtin_bit_cast(uint16_t, h);
}
__device__ __forceinline__ float bf2f(uint16_t b) {
  return __uint_as_float(((uint32_t)b) << 16);
}
__device__ __forceinline__ v8f wmma_bf16(v16bf a, v16bf b, v8f c) {
  return __builtin_amdgcn_wmma_f32_16x16x32_bf16(false, a, false, b, (short)0, c,
                                                 false, false);
}
__device__ __forceinline__ v16bf join_frag(v4u lo, v4u hi) {
  v8u u;
  u[0] = lo[0]; u[1] = lo[1]; u[2] = lo[2]; u[3] = lo[3];
  u[4] = hi[0]; u[5] = hi[1]; u[6] = hi[2]; u[7] = hi[3];
  return __builtin_bit_cast(v16bf, u);
}
// A fragment (16x32 bf16): two 16B chunks at pair offsets kt*16+hi*4 and kt*16+8+hi*4
__device__ __forceinline__ v16bf a_frag_from(const uint32_t* rowbase, int kt, int hi) {
  v4u lo = *(const v4u*)(rowbase + kt * 16 + hi * 4);
  v4u hq = *(const v4u*)(rowbase + kt * 16 + 8 + hi * 4);
  return join_frag(lo, hq);
}

// ---------- kernel 1: weight prep (f32 -> packed bf16 pairs), bias combine ----------
__global__ void prep_kernel(const float* __restrict__ Wih, const float* __restrict__ Whh,
                            const float* __restrict__ bih, const float* __restrict__ bhh,
                            uint32_t* __restrict__ wih_p, uint32_t* __restrict__ whh_p,
                            float* __restrict__ bcomb) {
  const int NIH = GATES * (EMB / 2);   // 65536
  const int NHH = GATES * (HID / 2);   // 131072
  int i = blockIdx.x * blockDim.x + threadIdx.x;
  if (i < NIH) {
    wih_p[i] = cvt_pk(Wih[2 * i], Wih[2 * i + 1]);       // row-major pairs
  }
  int j = i - NIH;
  if (j >= 0 && j < NHH) {
    whh_p[j] = cvt_pk(Whh[2 * j], Whh[2 * j + 1]);
  }
  int k = i - NIH - NHH;
  if (k >= 0 && k < GATES) bcomb[k] = bih[k] + bhh[k];
}

// ---------- kernel 2: xg[T][1024] = gather(E)@W_ih^T + b  (WMMA bf16) ----------
__global__ void xg_kernel(const int* __restrict__ sentence, const float* __restrict__ emb,
                          const uint32_t* __restrict__ wih_p, const float* __restrict__ bcomb,
                          float* __restrict__ xg) {
  const int lane = threadIdx.x & 31;
  const int w    = threadIdx.x >> 5;            // 8 waves / block
  const int t0   = blockIdx.x * 16;             // 16 timesteps per block
  const int n0   = blockIdx.y * 128 + w * 16;   // 16 gate columns per wave
  const int mrow = lane & 15;
  const int hi   = (lane >= 16) ? 1 : 0;

  const int tok = sentence[t0 + mrow];
  const float*    erow = emb + (size_t)tok * EMB;
  const uint32_t* brow = wih_p + (size_t)(n0 + mrow) * (EMB / 2);

  v8f acc = {};
#pragma unroll
  for (int kt = 0; kt < 4; ++kt) {              // K = 128 = 4 * 32
    // A: gather 16 f32 (four b128 loads) and pack to bf16
    float4 f0 = *(const float4*)(erow + kt * 32 + hi * 8);
    float4 f1 = *(const float4*)(erow + kt * 32 + hi * 8 + 4);
    float4 f2 = *(const float4*)(erow + kt * 32 + 16 + hi * 8);
    float4 f3 = *(const float4*)(erow + kt * 32 + 16 + hi * 8 + 4);
    v8u au;
    au[0] = cvt_pk(f0.x, f0.y); au[1] = cvt_pk(f0.z, f0.w);
    au[2] = cvt_pk(f1.x, f1.y); au[3] = cvt_pk(f1.z, f1.w);
    au[4] = cvt_pk(f2.x, f2.y); au[5] = cvt_pk(f2.z, f2.w);
    au[6] = cvt_pk(f3.x, f3.y); au[7] = cvt_pk(f3.z, f3.w);
    // B: two 16B chunks at pair offsets kt*16+hi*4 and kt*16+8+hi*4 (two b128 loads)
    v4u b0 = *(const v4u*)(brow + kt * 16 + hi * 4);
    v4u b1 = *(const v4u*)(brow + kt * 16 + 8 + hi * 4);
    acc = wmma_bf16(__builtin_bit_cast(v16bf, au), join_frag(b0, b1), acc);
  }
  const float bc = bcomb[n0 + mrow];
#pragma unroll
  for (int r = 0; r < 8; ++r) {
    int t = t0 + r + 8 * hi;                    // D: VGPR r -> rows r / r+8
    xg[(size_t)t * GATES + n0 + mrow] = acc[r] + bc;
  }
}

// ---------- kernel 3: persistent single-WGP LSTM recurrence ----------
__global__ __launch_bounds__(512)
void lstm_kernel(const uint32_t* __restrict__ whh_p, const float* __restrict__ xg,
                 uint16_t* __restrict__ hbf) {
  extern __shared__ char smem[];
  uint32_t* lds_whh = (uint32_t*)smem;                               // [512][PITCH]
  float*    gates   = (float*)(smem + 512 * PITCH * 4);              // [1024]
  uint32_t* h_pairs = (uint32_t*)(smem + 512 * PITCH * 4 + 4096);    // [128] bf16 pairs
  uint16_t* h_half  = (uint16_t*)h_pairs;

  const int tid  = threadIdx.x;
  const int lane = tid & 31;
  const int w    = tid >> 5;                    // 16 waves
  const int mrow = lane & 15;
  const int hi   = (lane >= 16) ? 1 : 0;

  // ---- stage g,o weights (rows 512..1023) into LDS with async b128 copies ----
  for (int chunk = tid; chunk < 512 * 8; chunk += 512) {   // 512 rows x 8 x 16B
    int r = chunk >> 3, c = chunk & 7;
    uint32_t ldsoff = (uint32_t)(uintptr_t)(lds_whh + r * PITCH + c * 4);
    const uint32_t* gp = whh_p + (size_t)(512 + r) * (HID / 2) + c * 4;
    asm volatile("global_load_async_to_lds_b128 %0, %1, off"
                 :: "v"(ldsoff), "v"(gp) : "memory");
  }
  if (tid < 128) h_pairs[tid] = 0u;             // h0 = 0
  asm volatile("s_wait_asynccnt 0" ::: "memory");

  // ---- preload i,f weight fragments (rows 0..511) into VGPRs ----
  v16bf a_if[2][8];
#pragma unroll
  for (int m = 0; m < 2; ++m) {
    const uint32_t* rowp = whh_p + (size_t)(32 * w + 16 * m + mrow) * (HID / 2);
#pragma unroll
    for (int kt = 0; kt < 8; ++kt) a_if[m][kt] = a_frag_from(rowp, kt, hi);
  }
  __syncthreads();

  float c_reg = 0.0f;                           // cell state owned by tid<256
  float x0 = 0.f, x1 = 0.f, x2 = 0.f, x3 = 0.f; // software-pipelined xg[t] row
  if (tid < HID) {
    x0 = xg[tid]; x1 = xg[256 + tid]; x2 = xg[512 + tid]; x3 = xg[768 + tid];
  }

  const uint32_t* lrow2 = lds_whh + (32 * w + mrow) * PITCH;        // tile 0: g rows (w<8) or o rows (w>=8)
  const uint32_t* lrow3 = lds_whh + (32 * w + 16 + mrow) * PITCH;   // tile 1: g rows (w<8) or o rows (w>=8)

  for (int t = 0; t < T_LEN; ++t) {
    v8f acc0 = {}, acc1 = {}, acc2 = {}, acc3 = {};
#pragma unroll
    for (int kt = 0; kt < 8; ++kt) {            // K = 256 = 8 * 32
      // B = h broadcast into all 16 columns: two b128 LDS loads
      v4u b0 = *(const v4u*)(h_pairs + kt * 16 + hi * 4);
      v4u b1 = *(const v4u*)(h_pairs + kt * 16 + 8 + hi * 4);
      v16bf b = join_frag(b0, b1);
      acc0 = wmma_bf16(a_if[0][kt], b, acc0);   // VGPR-resident i,f tiles
      acc1 = wmma_bf16(a_if[1][kt], b, acc1);
      acc2 = wmma_bf16(a_frag_from(lrow2, kt, hi), b, acc2);  // LDS-resident g,o
      acc3 = wmma_bf16(a_frag_from(lrow3, kt, hi), b, acc3);
    }
    // cols replicated: lane0 stores rows 0-7, lane16 rows 8-15 (b128 stores)
    if (lane == 0 || lane == 16) {
      int rb = 8 * hi;
      *(float4*)(gates + 32 * w + rb)            = make_float4(acc0[0], acc0[1], acc0[2], acc0[3]);
      *(float4*)(gates + 32 * w + rb + 4)        = make_float4(acc0[4], acc0[5], acc0[6], acc0[7]);
      *(float4*)(gates + 32 * w + 16 + rb)       = make_float4(acc1[0], acc1[1], acc1[2], acc1[3]);
      *(float4*)(gates + 32 * w + 16 + rb + 4)   = make_float4(acc1[4], acc1[5], acc1[6], acc1[7]);
      *(float4*)(gates + 512 + 32 * w + rb)      = make_float4(acc2[0], acc2[1], acc2[2], acc2[3]);
      *(float4*)(gates + 512 + 32 * w + rb + 4)  = make_float4(acc2[4], acc2[5], acc2[6], acc2[7]);
      *(float4*)(gates + 512 + 32 * w + 16 + rb)     = make_float4(acc3[0], acc3[1], acc3[2], acc3[3]);
      *(float4*)(gates + 512 + 32 * w + 16 + rb + 4) = make_float4(acc3[4], acc3[5], acc3[6], acc3[7]);
    }
    __syncthreads();
    if (tid < HID) {                            // activations, one hidden unit each
      float gi = gates[tid]       + x0;
      float gf = gates[256 + tid] + x1;
      float gg = gates[512 + tid] + x2;
      float go = gates[768 + tid] + x3;
      float iv = 1.f / (1.f + __expf(-gi));
      float fv = 1.f / (1.f + __expf(-gf));
      float gv = tanhf(gg);
      float ov = 1.f / (1.f + __expf(-go));
      c_reg = fv * c_reg + iv * gv;
      float h = ov * tanhf(c_reg);
      uint16_t hb = f2bf(h);
      h_half[tid] = hb;                         // next-step B operand
      hbf[(size_t)t * HID + tid] = hb;          // for output projection
      // prefetch xg[t+1]: loads overlap the entire next phase A
      int tn = (t + 1 < T_LEN) ? (t + 1) : t;
      const float* xn = xg + (size_t)tn * GATES;
      x0 = xn[tid]; x1 = xn[256 + tid]; x2 = xn[512 + tid]; x3 = xn[768 + tid];
    }
    __syncthreads();
  }
}

// ---------- kernel 4: tag projection + log_softmax (tiny) ----------
__global__ void out_kernel(const uint16_t* __restrict__ hbf, const float* __restrict__ Wout,
                           const float* __restrict__ bout, float* __restrict__ out) {
  int t = blockIdx.x * blockDim.x + threadIdx.x;
  if (t >= T_LEN) return;
  const uint16_t* hrow = hbf + (size_t)t * HID;
  float s[TAGS];
#pragma unroll
  for (int g = 0; g < TAGS; ++g) s[g] = bout[g];
  for (int k = 0; k < HID; ++k) {
    float hv = bf2f(hrow[k]);
#pragma unroll
    for (int g = 0; g < TAGS; ++g) s[g] += hv * Wout[g * HID + k];
  }
  float m = s[0];
#pragma unroll
  for (int g = 1; g < TAGS; ++g) m = fmaxf(m, s[g]);
  float se = 0.f;
#pragma unroll
  for (int g = 0; g < TAGS; ++g) se += __expf(s[g] - m);
  float lse = m + __logf(se);
#pragma unroll
  for (int g = 0; g < TAGS; ++g) out[(size_t)t * TAGS + g] = s[g] - lse;
}

// ---------- launcher ----------
extern "C" void kernel_launch(void* const* d_in, const int* in_sizes, int n_in,
                              void* d_out, int out_size, void* d_ws, size_t ws_size,
                              hipStream_t stream) {
  const int*   sentence = (const int*)  d_in[0];
  const float* emb      = (const float*)d_in[1];
  const float* Wih      = (const float*)d_in[2];
  const float* Whh      = (const float*)d_in[3];
  const float* bih      = (const float*)d_in[4];
  const float* bhh      = (const float*)d_in[5];
  const float* Wout     = (const float*)d_in[6];
  const float* bout     = (const float*)d_in[7];
  float* out = (float*)d_out;

  char* ws = (char*)d_ws;
  uint32_t* wih_p = (uint32_t*)(ws);                           // 256 KB
  uint32_t* whh_p = (uint32_t*)(ws + (256 << 10));             // 512 KB
  float*    bcomb = (float*)   (ws + (768 << 10));             // 4 KB
  uint16_t* hbf   = (uint16_t*)(ws + (772 << 10));             // 4 MB
  float*    xg    = (float*)   (ws + (772 << 10) + (4 << 20)); // 32 MB

  {
    int total = GATES * (EMB / 2) + GATES * (HID / 2) + GATES;
    prep_kernel<<<(total + 255) / 256, 256, 0, stream>>>(Wih, Whh, bih, bhh,
                                                         wih_p, whh_p, bcomb);
  }
  {
    dim3 grid(T_LEN / 16, GATES / 128);
    xg_kernel<<<grid, 256, 0, stream>>>(sentence, emb, wih_p, bcomb, xg);
  }
  {
    size_t lds_bytes = 512 * PITCH * 4 + 4096 + 512;           // ~269 KB of 320 KB
    lstm_kernel<<<1, 512, lds_bytes, stream>>>(whh_p, xg, hbf);
  }
  out_kernel<<<(T_LEN + 127) / 128, 128, 0, stream>>>(hbf, Wout, bout, out);
}
